// NoisyIBLayer_84911503442655
// MI455X (gfx1250) — matrix operand order repaired and checked by reading, not compiled, not run on hardware
//
#include <hip/hip_runtime.h>

// N=8192 rows, D=256 features. Core cost: G = X X^T (34.4 GFLOP), fused with
// row-wise double logsumexp so the 256MB dist matrix is never materialized.
#define N_ROWS 8192
#define D_DIM  256

typedef __attribute__((ext_vector_type(16))) __bf16 bf16x16;
typedef __attribute__((ext_vector_type(8)))  __bf16 bf16x8;
typedef __attribute__((ext_vector_type(8)))  float  f32x8;
// exact pointee type the async-LDS builtin expects (per hipcc diagnostic):
typedef int i32x4 __attribute__((vector_size(16)));

#define AS1 __attribute__((address_space(1)))
#define AS3 __attribute__((address_space(3)))

// CDNA5 async global->LDS copy (ASYNCcnt path). Guarded: falls back to the
// (already working) VGPR round-trip staging if the builtin is absent.
#if defined(__has_builtin)
# if __has_builtin(__builtin_amdgcn_global_load_async_to_lds_b128)
#  define USE_ASYNC_LDS 1
# endif
#endif
#ifndef USE_ASYNC_LDS
# define USE_ASYNC_LDS 0
#endif

__device__ __forceinline__ void wait_async_lds() {
#if USE_ASYNC_LDS
# if defined(__has_builtin) && __has_builtin(__builtin_amdgcn_s_wait_asynccnt)
    __builtin_amdgcn_s_wait_asynccnt(0);
# else
    asm volatile("s_wait_asynccnt 0x0" ::: "memory");
# endif
#endif
}

__device__ __forceinline__ float softplus_f(float x) {
    return (x > 20.f) ? x : log1pf(__expf(x));
}

// ---------------------------------------------------------------------------
// Kernel A: streaming pass. One 256-thread block per row.
//  - out = x + noise*sqrt(nv)
//  - xb  = bf16(x)            (feeds the WMMA GEMM)
//  - sq[row] = sum_j x[row,j]^2  (needed for dist AND for vIxt)
//  - zero the S accumulators (must be re-zeroed every launch: determinism)
// ---------------------------------------------------------------------------
__global__ void __launch_bounds__(256)
prep_kernel(const float* __restrict__ x, const float* __restrict__ noise,
            const float* __restrict__ phi, float* __restrict__ out,
            __bf16* __restrict__ xb, float* __restrict__ sq,
            float* __restrict__ S_l, float* __restrict__ S_u) {
    __shared__ float red[256];
    const int row = blockIdx.x;
    const int t = threadIdx.x;
    const float nv = softplus_f(phi[0]);
    const float sdev = sqrtf(nv);
    const float v = x[row * D_DIM + t];
    out[row * D_DIM + t] = v + noise[row * D_DIM + t] * sdev;
    xb[row * D_DIM + t] = (__bf16)v;
    red[t] = v * v;
    __syncthreads();
    for (int s = 128; s > 0; s >>= 1) {
        if (t < s) red[t] += red[t + s];
        __syncthreads();
    }
    if (t == 0) { sq[row] = red[0]; S_l[row] = 0.f; S_u[row] = 0.f; }
}

// Stage one 64x256 bf16 B tile (32KB) into LDS. 256 threads: 4 threads/row,
// 128B (= 8 x b128) each. Async path: pure DMA, no VGPR round-trip.
__device__ __forceinline__ void stage_tile(const __bf16* __restrict__ xb,
                                           __bf16* dst_lds, int j0, int t) {
    const int row = t >> 2;
    const int off = (t & 3) * 64;
    const __bf16* src = xb + (j0 + row) * D_DIM + off;
    __bf16* dst = dst_lds + row * D_DIM + off;
#if USE_ASYNC_LDS
#pragma unroll
    for (int c = 0; c < 8; ++c)
        __builtin_amdgcn_global_load_async_to_lds_b128(
            (AS1 i32x4*)(src + c * 8), (AS3 i32x4*)(dst + c * 8), 0, 0);
#else
    const bf16x8* s8 = (const bf16x8*)src;
    bf16x8* d8 = (bf16x8*)dst;
#pragma unroll
    for (int c = 0; c < 8; ++c) d8[c] = s8[c];
#endif
}

// ---------------------------------------------------------------------------
// Kernel B: fused Gram-matrix + logsumexp partial sums.
// Grid: (64 row-blocks of 128) x (8 column-chunks of 1024). 256 thr = 8 waves.
// Each wave owns 16 output rows: A fragments (16x256 bf16 = 64 VGPRs) stay in
// registers for the whole J loop. B tiles (64x256 bf16, double-buffered in
// 2x32KB LDS) streamed with async global->LDS DMA: tile jt+1 is in flight
// while WMMAs consume tile jt. 8x4 unrolled v_wmma_f32_16x16x32_bf16 per wave
// per J-tile. Epilogue: dist = sq_i + sq_j - 2g; e_l = exp(-dist/(8v));
// e_u = e_l^4 (since 1/(2v) = 4/(8v)) -> only ONE v_exp per element.
// ---------------------------------------------------------------------------
__global__ void __launch_bounds__(256)
gram_lse_kernel(const __bf16* __restrict__ xb, const float* __restrict__ sq,
                const float* __restrict__ phi, float* __restrict__ S_l,
                float* __restrict__ S_u) {
    __shared__ __bf16 Btile[2][64 * D_DIM];  // 2 x 32 KB ping-pong

    const int lane = threadIdx.x & 31;
    const int wave = threadIdx.x >> 5;
    const int half = lane >> 4;   // ISA bf16 A/B layouts split K by lane-half
    const int l15  = lane & 15;
    const int mrow = blockIdx.x * 128 + wave * 16;  // this wave's 16 rows
    const int cbase = blockIdx.y * 1024;            // this WG's column chunk

    const float nv = softplus_f(phi[0]);
    const float inv8 = 1.f / (8.f * nv);

    // --- A fragments: 16 rows x 256 K, ISA layout:
    // lanes 0-15 hold M=lane,    K = kc*32 + {0..7, 16..23}
    // lanes 16-31 hold M=lane-16, K = kc*32 + {8..15, 24..31}
    bf16x16 afrag[8];
    {
        const __bf16* arow = xb + (mrow + l15) * D_DIM;
#pragma unroll
        for (int kc = 0; kc < 8; ++kc) {
            bf16x8 a0 = *(const bf16x8*)(arow + kc * 32 + half * 8);
            bf16x8 a1 = *(const bf16x8*)(arow + kc * 32 + 16 + half * 8);
            afrag[kc] = __builtin_shufflevector(a0, a1, 0, 1, 2, 3, 4, 5, 6, 7,
                                                8, 9, 10, 11, 12, 13, 14, 15);
        }
    }

    // row norms for this wave's rows (C layout: VGPR r, half h -> M = r + 8h)
    float si[8];
#pragma unroll
    for (int r = 0; r < 8; ++r) si[r] = sq[mrow + r + 8 * half];

    float pl[8], pu[8];
#pragma unroll
    for (int r = 0; r < 8; ++r) { pl[r] = 0.f; pu[r] = 0.f; }

    // prologue: kick off DMA for the first tile
    stage_tile(xb, &Btile[0][0], cbase, threadIdx.x);

    for (int jt = 0; jt < 16; ++jt) {
        const int cur = jt & 1;
        const int j0 = cbase + jt * 64;

        wait_async_lds();     // my DMA writes into Btile[cur] have landed
        __syncthreads();      // whole tile present; prev reads of other buf done

        // issue DMA for the next tile into the other buffer (overlaps compute)
        if (jt + 1 < 16)
            stage_tile(xb, &Btile[cur ^ 1][0], cbase + (jt + 1) * 64, threadIdx.x);

        const __bf16* bt = &Btile[cur][0];

        f32x8 acc[4];
#pragma unroll
        for (int tc = 0; tc < 4; ++tc)
            acc[tc] = (f32x8){0.f, 0.f, 0.f, 0.f, 0.f, 0.f, 0.f, 0.f};

#pragma unroll
        for (int kc = 0; kc < 8; ++kc) {
            bf16x16 a = afrag[kc];
#pragma unroll
            for (int tc = 0; tc < 4; ++tc) {
                // B (K x N): lane holds column n = tc*16 + l15 = row n of X_J,
                // K = kc*32 + half*8 + {0..7, 16..23} (same as A)
                const __bf16* bp = bt + (tc * 16 + l15) * D_DIM + kc * 32 + half * 8;
                bf16x8 b0 = *(const bf16x8*)bp;
                bf16x8 b1 = *(const bf16x8*)(bp + 16);
                bf16x16 b = __builtin_shufflevector(
                    b0, b1, 0, 1, 2, 3, 4, 5, 6, 7, 8, 9, 10, 11, 12, 13, 14, 15);
                acc[tc] = __builtin_amdgcn_wmma_f32_16x16x32_bf16(
                    false, a, false, b, (short)0, acc[tc], false, false);
            }
        }

        // epilogue: exp accumulation (TRANS ops co-execute with XDL WMMA)
#pragma unroll
        for (int tc = 0; tc < 4; ++tc) {
            const float sj = sq[j0 + tc * 16 + l15];
#pragma unroll
            for (int r = 0; r < 8; ++r) {
                float g = acc[tc][r];
                float d = fmaxf(si[r] + sj - 2.f * g, 0.f);
                float el = __expf(-d * inv8);  // lower bound: denom = 8v
                float e2 = el * el;
                pl[r] += el;
                pu[r] += e2 * e2;              // upper bound: denom = 2v -> el^4
            }
        }
        __syncthreads();  // all reads of Btile[cur] done before it is re-filled
    }

    // reduce across the 16 lanes of each half, then 2 atomics per row
#pragma unroll
    for (int r = 0; r < 8; ++r) {
        float a = pl[r], b = pu[r];
#pragma unroll
        for (int m = 1; m < 16; m <<= 1) {
            a += __shfl_xor(a, m, 32);
            b += __shfl_xor(b, m, 32);
        }
        if (l15 == 0) {
            const int i = mrow + r + 8 * half;
            atomicAdd(&S_l[i], a);
            atomicAdd(&S_u[i], b);
        }
    }
}

// ---------------------------------------------------------------------------
// Kernel C: scalar finalize. Ixt_* = log N - mean_i log S_*  (const - H = logN)
// vIxt = D*(0.5 log(pv/nv) + nv/(2pv) - 0.5) + sum(x^2)/(2 pv N)
// ---------------------------------------------------------------------------
__global__ void __launch_bounds__(256)
finalize_kernel(const float* __restrict__ S_l, const float* __restrict__ S_u,
                const float* __restrict__ sq, const float* __restrict__ phi,
                const float* __restrict__ prior, float* __restrict__ out_sc) {
    __shared__ float rl[256], ru[256], rs[256];
    const int t = threadIdx.x;
    float al = 0.f, au = 0.f, as_ = 0.f;
    for (int i = t; i < N_ROWS; i += 256) {
        al += logf(S_l[i]);
        au += logf(S_u[i]);
        as_ += sq[i];
    }
    rl[t] = al; ru[t] = au; rs[t] = as_;
    __syncthreads();
    for (int s = 128; s > 0; s >>= 1) {
        if (t < s) { rl[t] += rl[t + s]; ru[t] += ru[t + s]; rs[t] += rs[t + s]; }
        __syncthreads();
    }
    if (t == 0) {
        const float nv = softplus_f(phi[0]);
        const float pv = prior[0];
        const float logN = logf((float)N_ROWS);
        out_sc[0] = logN - rl[0] / (float)N_ROWS;  // Ixt_lb (denom 8v)
        out_sc[1] = logN - ru[0] / (float)N_ROWS;  // Ixt    (denom 2v)
        out_sc[2] = (float)D_DIM * (0.5f * logf(pv / nv) + nv / (2.f * pv) - 0.5f)
                    + rs[0] / (2.f * pv * (float)N_ROWS);
    }
}

extern "C" void kernel_launch(void* const* d_in, const int* in_sizes, int n_in,
                              void* d_out, int out_size, void* d_ws, size_t ws_size,
                              hipStream_t stream) {
    const float* x     = (const float*)d_in[0];
    const float* noise = (const float*)d_in[1];
    const float* phi   = (const float*)d_in[2];
    const float* prior = (const float*)d_in[3];
    float* out = (float*)d_out;
    float* out_scalars = out + (size_t)N_ROWS * D_DIM;

    // workspace layout: bf16 X copy (4MB) | sq | S_l | S_u
    __bf16* xb = (__bf16*)d_ws;
    float* sq  = (float*)((char*)d_ws + (size_t)N_ROWS * D_DIM * sizeof(__bf16));
    float* S_l = sq + N_ROWS;
    float* S_u = S_l + N_ROWS;

    prep_kernel<<<N_ROWS, 256, 0, stream>>>(x, noise, phi, out, xb, sq, S_l, S_u);
    gram_lse_kernel<<<dim3(64, 8), 256, 0, stream>>>(xb, sq, phi, S_l, S_u);
    finalize_kernel<<<1, 256, 0, stream>>>(S_l, S_u, sq, phi, prior, out_scalars);
}
